// IPA_80711025426896
// MI455X (gfx1250) — compile-verified, matrix operand order checked by reading, not run on hardware
//
#include <hip/hip_runtime.h>
#include <math.h>

typedef __attribute__((ext_vector_type(16))) _Float16 v16h;
typedef __attribute__((ext_vector_type(8)))  _Float16 v8h;
typedef __attribute__((ext_vector_type(8)))  float    v8f;
typedef __attribute__((ext_vector_type(4)))  float    v4f;
typedef __attribute__((ext_vector_type(4)))  unsigned int u32x4;
typedef __attribute__((ext_vector_type(8)))  int      i32x8;
typedef __attribute__((ext_vector_type(4)))  int      i32x4;

// Problem constants (N=1)
constexpr int LSEQ = 512;
constexpr int CN   = 384;
constexpr int CE   = 128;
constexpr int NG   = 12;
constexpr int PPG  = 16;
constexpr int NPQ  = 4;
constexpr int NPV  = 8;
constexpr int PW   = 1152;
constexpr int FEATW = 2112;
constexpr long PLANE = (long)LSEQ * LSEQ; // 262144

// 8 contiguous elements -> f16 vector (16B vector loads; f32 converts).
__device__ inline v8h ldcvt8(const _Float16* __restrict__ p) { return *(const v8h*)p; }
__device__ inline v8h ldcvt8(const float* __restrict__ p) {
  const v4f a = *(const v4f*)p;
  const v4f b = *(const v4f*)(p + 4);
  v8h r;
#pragma unroll
  for (int i = 0; i < 4; ++i) { r[i] = (_Float16)a[i]; r[i + 4] = (_Float16)b[i]; }
  return r;
}

// CDNA5 16-bit fragment: lane's 16 halfs = rows [khi..khi+7] and [khi+16..khi+23]
// of a 32-wide row -> two contiguous b128 LDS reads.
__device__ inline v16h frag32(const _Float16* __restrict__ p) {
  const v8h a = *(const v8h*)p;
  const v8h b = *(const v8h*)(p + 16);
  return __builtin_shufflevector(a, b, 0,1,2,3,4,5,6,7,8,9,10,11,12,13,14,15);
}

// ---------------------------------------------------------------------------
// Generic strided WMMA GEMM. Requirements (all call sites satisfy them):
//   * A is K-contiguous (stride 1 along k), rows 16B aligned, K % 32 == 0
//   * B: if BKC, K-contiguous with row stride bs (per n); else n-contiguous
//     with row stride bs (per k)
//   * loads may run to the 16-padded M/N (buffers padded); C store is guarded
// 256 threads = 8 wave32; wave (wm,wn) owns one 16x16 tile; B kept transposed
// in LDS so both fragments are pure b128 reads.
// ---------------------------------------------------------------------------
template<int TM, int TN, typename TA, typename TB, typename TC, bool ACC, bool BKC>
__global__ __launch_bounds__(256) void gemm_wmma_k(
    const TA* __restrict__ A, long am, long bsA,
    const TB* __restrict__ B, long bs, long bsB,
    TC* __restrict__ C, long cm, long cn, long bsC,
    const float* __restrict__ biasN, int M, int N, int K)
{
  static_assert(TM * TN == 8, "8 waves per block");
  constexpr int BM = TM * 16, BN = TN * 16;
  __shared__ alignas(16) _Float16 lA[BM * 32];
  __shared__ alignas(16) _Float16 lBt[BN * 32];   // transposed: [n][k]

  const int tid  = threadIdx.x;
  const int lane = tid & 31;
  const int wave = tid >> 5;
  const int wm = wave / TN;
  const int wn = wave % TN;
  const int m0 = blockIdx.y * BM;
  const int n0 = blockIdx.x * BN;
  const long zA = (long)blockIdx.z * bsA;
  const long zB = (long)blockIdx.z * bsB;
  const long zC = (long)blockIdx.z * bsC;
  const int row = lane & 15;
  const int khi = (lane >> 4) << 3;

  v8f acc = {};
  for (int k0 = 0; k0 < K; k0 += 32) {
    // A tile: BM x 32, b128 chunks.
    for (int idx = tid; idx < BM * 4; idx += 256) {
      const int m = idx >> 2, kc = (idx & 3) << 3;
      *(v8h*)(lA + m * 32 + kc) = ldcvt8(A + zA + (long)(m0 + m) * am + (k0 + kc));
    }
    // B tile -> transposed LDS.
    if (BKC) {
      for (int idx = tid; idx < BN * 4; idx += 256) {
        const int n = idx >> 2, kc = (idx & 3) << 3;
        *(v8h*)(lBt + n * 32 + kc) = ldcvt8(B + zB + (long)(n0 + n) * bs + (k0 + kc));
      }
    } else {
      for (int idx = tid; idx < BN * 4; idx += 256) {
        const int ng = idx % (BN / 8);
        const int kk = idx / (BN / 8);
        const v8h h = ldcvt8(B + zB + (long)(k0 + kk) * bs + (n0 + ng * 8));
        const int nb = ng * 8;
#pragma unroll
        for (int u = 0; u < 8; ++u) lBt[(nb + u) * 32 + kk] = h[u];
      }
    }
    if (k0 + 32 < K)
      __builtin_prefetch(A + zA + (long)(m0 + wave * (BM / 8)) * am + (k0 + 32), 0, 1);
    __syncthreads();

    const v16h af = frag32(lA  + (wm * 16 + row) * 32 + khi);
    const v16h bf = frag32(lBt + (wn * 16 + row) * 32 + khi);
    acc = __builtin_amdgcn_wmma_f32_16x16x32_f16(false, af, false, bf,
                                                 (short)0, acc, false, false);
    __syncthreads();
  }

  const int n  = lane & 15;
  const int mb = (lane >> 4) << 3;
#pragma unroll
  for (int r = 0; r < 8; ++r) {
    const int gm = m0 + wm * 16 + mb + r;
    const int gn = n0 + wn * 16 + n;
    if (gm < M && gn < N) {
      float v = acc[r];
      const long co = zC + (long)gm * cm + (long)gn * cn;
      if (ACC) v += (float)C[co];
      if (biasN) v += biasN[gn];
      C[co] = (TC)v;
    }
  }
}

// ---------------------------------------------------------------------------
// Edge pass 1 with the Tensor Data Mover: each block DMAs a 128x128 f32 Edge
// tile into LDS (tensor_load_to_lds, TENSORcnt), then
//  (a) 8 waves run 4 WMMA k-steps against the padded/transposed Wbias tile:
//      S[g, i, j] = sum_c Edge[i, j, c] * Wbias[c, g]
//  (b) the tile is written back out as f16, transposed to Edge_t[i][c][j],
//      so the O_Edge pass gets K(=j)-contiguous operands.
// ---------------------------------------------------------------------------
__global__ __launch_bounds__(256) void bias_edge_tdm_k(
    const float* __restrict__ Edge, const _Float16* __restrict__ Wb,
    float* __restrict__ S, _Float16* __restrict__ Edge_t)
{
  __shared__ alignas(16) float    tA[128 * 128];   // 64 KB Edge tile (j x c)
  __shared__ alignas(16) _Float16 tB[16 * 128];    // Wbias transposed [g][c]
  const int i  = blockIdx.y;
  const int j0 = blockIdx.x * 128;
  const int tid = threadIdx.x, lane = tid & 31;
  const int row = lane & 15, khi = (lane >> 4) << 3;

#if defined(__AMDGCN__) && __has_builtin(__builtin_amdgcn_tensor_load_to_lds)
  if (__builtin_amdgcn_readfirstlane(tid >> 5) == 0) {
    // D# group0: count=1 | lds_addr | global_addr(57b) | type=2
    const unsigned long long ga =
        (unsigned long long)(const void*)(Edge + ((long)i * LSEQ + j0) * CE);
    const unsigned int la = (unsigned int)(unsigned long long)(const void*)&tA[0];
    const u32x4 g0 = { 1u, la, (unsigned int)ga,
                       (unsigned int)((ga >> 32) | 0x80000000u) };
    // D# group1: data_size=4B; tensor_dim0=128, tensor_dim1=128,
    // tile_dim0=128, tile_dim1=128, tensor_dim0_stride=128
    const i32x8 g1 = { 0x00020000, (int)(128u << 16), (int)(128u << 16),
                       (int)(128u << 16), 128, 128, 0, 0 };
    const i32x4 gz = { 0, 0, 0, 0 };
#if __clang_major__ >= 23
    const i32x8 gz8 = { 0, 0, 0, 0, 0, 0, 0, 0 };
    __builtin_amdgcn_tensor_load_to_lds(g0, g1, gz, gz, gz8, 0);
#else
    __builtin_amdgcn_tensor_load_to_lds(g0, g1, gz, gz, 0);
#endif
  }
#else
  for (int idx = tid; idx < 128 * 32; idx += 256) {   // cooperative fallback
    const int r = idx >> 5, c = (idx & 31) << 2;
    *(v4f*)(tA + r * 128 + c) =
        *(const v4f*)(Edge + ((long)i * LSEQ + j0 + r) * CE + c);
  }
#endif

  // Wbias (padded to 16 cols) -> transposed LDS [g][c]
  for (int idx = tid; idx < 256; idx += 256) {
    const int n = idx >> 4, kc = (idx & 15) << 3;
    v8h h;
#pragma unroll
    for (int u = 0; u < 8; ++u) h[u] = Wb[(kc + u) * 16 + n];
    *(v8h*)(tB + n * 128 + kc) = h;
  }

#if defined(__AMDGCN__) && __has_builtin(__builtin_amdgcn_s_wait_tensorcnt)
  __builtin_amdgcn_s_wait_tensorcnt(0);
#endif
  __syncthreads();

  // (a) bias WMMA
  const int wave = tid >> 5;
  v8f acc = {};
#pragma unroll
  for (int ks = 0; ks < 4; ++ks) {
    const int k0 = ks * 32;
    const float* par = tA + (wave * 16 + row) * 128 + k0 + khi;
    const v4f f0 = *(const v4f*)par;
    const v4f f1 = *(const v4f*)(par + 4);
    const v4f f2 = *(const v4f*)(par + 16);
    const v4f f3 = *(const v4f*)(par + 20);
    v16h af;
#pragma unroll
    for (int u = 0; u < 4; ++u) {
      af[u]      = (_Float16)f0[u];
      af[u + 4]  = (_Float16)f1[u];
      af[u + 8]  = (_Float16)f2[u];
      af[u + 12] = (_Float16)f3[u];
    }
    const v16h bf = frag32(tB + row * 128 + k0 + khi);
    acc = __builtin_amdgcn_wmma_f32_16x16x32_f16(false, af, false, bf,
                                                 (short)0, acc, false, false);
  }
  const int g = lane & 15, mb = (lane >> 4) << 3;
  if (g < NG) {
#pragma unroll
    for (int r = 0; r < 8; ++r)
      S[(long)g * PLANE + (long)i * LSEQ + j0 + wave * 16 + mb + r] = acc[r];
  }

  // (b) transposed f16 write-out: Edge_t[i][c][j0+jj]
  for (int idx = tid; idx < 128 * 16; idx += 256) {
    const int c  = idx >> 4;
    const int jg = (idx & 15) << 3;
    v8h h;
#pragma unroll
    for (int u = 0; u < 8; ++u) h[u] = (_Float16)tA[(jg + u) * 128 + c];
    *(v8h*)(Edge_t + ((long)i * CE + c) * LSEQ + j0 + jg) = h;
  }
}

// ---------------------------------------------------------------------------
// Small helper kernels
// ---------------------------------------------------------------------------
__global__ void calc_cg_k(const float* __restrict__ gamma,
                          float* __restrict__ cg, float* __restrict__ s2cg)
{
  const int g = threadIdx.x;
  if (g >= NG) return;
  const float sp = log1pf(__expf(gamma[g]));
  const float WC = sqrtf(2.f / (9.f * (float)NPQ));
  const float c  = sp * WC * 0.5f;
  cg[g]   = c;
  s2cg[g] = sqrtf(2.f * c);
}

__global__ void cvt_k(const float* __restrict__ src, _Float16* __restrict__ dst, int n)
{
  for (int i = blockIdx.x * blockDim.x + threadIdx.x; i < n; i += gridDim.x * blockDim.x)
    dst[i] = (_Float16)src[i];
}

__global__ void pack_wall_k(const float* __restrict__ Wq, const float* __restrict__ Wk,
                            const float* __restrict__ Wv, const float* __restrict__ Wqv,
                            const float* __restrict__ Wkv, const float* __restrict__ Wvv,
                            _Float16* __restrict__ Wall)
{
  const int n = CN * PW;
  for (int i = blockIdx.x * blockDim.x + threadIdx.x; i < n; i += gridDim.x * blockDim.x) {
    const int r = i / PW, c = i % PW;
    float v;
    if      (c < 192) v = Wq [r * 192 + c];
    else if (c < 384) v = Wk [r * 192 + (c - 192)];
    else if (c < 576) v = Wv [r * 192 + (c - 384)];
    else if (c < 720) v = Wqv[r * 144 + (c - 576)];
    else if (c < 864) v = Wkv[r * 144 + (c - 720)];
    else              v = Wvv[r * 288 + (c - 864)];
    Wall[i] = (_Float16)v;
  }
}

__global__ void pack_wbias_k(const float* __restrict__ Wb, _Float16* __restrict__ d)
{
  const int i = blockIdx.x * blockDim.x + threadIdx.x;
  if (i >= CE * 16) return;
  const int r = i / 16, c = i % 16;
  d[i] = (c < NG) ? (_Float16)Wb[r * NG + c] : (_Float16)0;
}

__global__ void frame_pack_k(const float* __restrict__ P,
                             const float* __restrict__ rots,
                             const float* __restrict__ trans,
                             const float* __restrict__ cg,
                             const float* __restrict__ s2cg,
                             _Float16* __restrict__ QH, _Float16* __restrict__ KH,
                             _Float16* __restrict__ VA,
                             float* __restrict__ sqnq, float* __restrict__ sqnk)
{
  const int idx = blockIdx.x * blockDim.x + threadIdx.x;
  if (idx >= LSEQ * NG) return;
  const int i = idx / NG, g = idx % NG;
  const float* R = rots + i * 9;
  const float tx = trans[i * 3 + 0] * 0.1f;
  const float ty = trans[i * 3 + 1] * 0.1f;
  const float tz = trans[i * 3 + 2] * 0.1f;
  const float* Pr = P + (long)i * PW;
  _Float16* qrow = QH + (long)i * (NG * 32) + g * 32;
  _Float16* krow = KH + (long)i * (NG * 32) + g * 32;
  _Float16* vrow = VA + (long)i * (NG * 48) + g * 48;
  const float s2 = s2cg[g], cgv = cg[g];

  for (int c = 0; c < PPG; ++c) {
    qrow[c] = (_Float16)(Pr[0   + g * PPG + c] * 0.25f);  // 1/sqrt(ppg)
    krow[c] = (_Float16)(Pr[192 + g * PPG + c]);
    vrow[c] = (_Float16)(Pr[384 + g * PPG + c]);
  }
  float aq = 0.f, ak = 0.f;
  for (int p = 0; p < NPQ; ++p) {
    const float* q3 = Pr + 576 + (g * NPQ + p) * 3;
    float fx = R[0]*q3[0] + R[1]*q3[1] + R[2]*q3[2] + tx;
    float fy = R[3]*q3[0] + R[4]*q3[1] + R[5]*q3[2] + ty;
    float fz = R[6]*q3[0] + R[7]*q3[1] + R[8]*q3[2] + tz;
    qrow[16 + p*3 + 0] = (_Float16)(s2 * fx);
    qrow[16 + p*3 + 1] = (_Float16)(s2 * fy);
    qrow[16 + p*3 + 2] = (_Float16)(s2 * fz);
    aq += fx*fx + fy*fy + fz*fz;
    const float* k3 = Pr + 720 + (g * NPQ + p) * 3;
    fx = R[0]*k3[0] + R[1]*k3[1] + R[2]*k3[2] + tx;
    fy = R[3]*k3[0] + R[4]*k3[1] + R[5]*k3[2] + ty;
    fz = R[6]*k3[0] + R[7]*k3[1] + R[8]*k3[2] + tz;
    krow[16 + p*3 + 0] = (_Float16)(s2 * fx);
    krow[16 + p*3 + 1] = (_Float16)(s2 * fy);
    krow[16 + p*3 + 2] = (_Float16)(s2 * fz);
    ak += fx*fx + fy*fy + fz*fz;
  }
  for (int c = 28; c < 32; ++c) { qrow[c] = (_Float16)0; krow[c] = (_Float16)0; }
  sqnq[idx] = cgv * aq;
  sqnk[idx] = cgv * ak;

  for (int p = 0; p < NPV; ++p) {
    const float* v3 = Pr + 864 + (g * NPV + p) * 3;
    const float fx = R[0]*v3[0] + R[1]*v3[1] + R[2]*v3[2] + tx;
    const float fy = R[3]*v3[0] + R[4]*v3[1] + R[5]*v3[2] + ty;
    const float fz = R[6]*v3[0] + R[7]*v3[1] + R[8]*v3[2] + tz;
    vrow[16 + p*3 + 0] = (_Float16)fx;
    vrow[16 + p*3 + 1] = (_Float16)fy;
    vrow[16 + p*3 + 2] = (_Float16)fz;
  }
  for (int c = 40; c < 48; ++c) vrow[c] = (_Float16)0;
}

__global__ __launch_bounds__(256) void softmax_k(const float* __restrict__ S,
                                                 const float* __restrict__ sqnq,
                                                 const float* __restrict__ sqnk,
                                                 _Float16* __restrict__ attn)
{
  constexpr float WL = 0.57735026919f; // sqrt(1/3)
  const int b = blockIdx.x;
  const int g = b >> 9, i = b & 511;
  const long base = ((long)g << 18) + ((long)i << 9);
  const int t = threadIdx.x;
  const float qn = sqnq[i * NG + g];
  const float x0 = WL * (S[base + t]       - qn - sqnk[(t)       * NG + g]);
  const float x1 = WL * (S[base + t + 256] - qn - sqnk[(t + 256) * NG + g]);

  __shared__ float redm[8], reds[8];
  const int lane = t & 31, wv = t >> 5;

  float m = fmaxf(x0, x1);
  for (int o = 16; o; o >>= 1) m = fmaxf(m, __shfl_xor(m, o, 32));
  if (lane == 0) redm[wv] = m;
  __syncthreads();
  m = redm[0];
#pragma unroll
  for (int w = 1; w < 8; ++w) m = fmaxf(m, redm[w]);

  const float e0 = __expf(x0 - m), e1 = __expf(x1 - m);
  float s = e0 + e1;
  for (int o = 16; o; o >>= 1) s += __shfl_xor(s, o, 32);
  if (lane == 0) reds[wv] = s;
  __syncthreads();
  s = reds[0];
#pragma unroll
  for (int w = 1; w < 8; ++w) s += reds[w];

  const float inv = 1.f / s;
  attn[base + t]       = (_Float16)(e0 * inv);
  attn[base + t + 256] = (_Float16)(e1 * inv);
}

__global__ void finalize_k(const float* __restrict__ Onv,
                           const float* __restrict__ rots,
                           const float* __restrict__ trans,
                           _Float16* __restrict__ feat)
{
  const int idx = blockIdx.x * blockDim.x + threadIdx.x;
  if (idx >= LSEQ * NG) return;
  const int i = idx / NG, g = idx % NG;
  const float* R = rots + i * 9;
  const float tx = trans[i*3+0]*0.1f, ty = trans[i*3+1]*0.1f, tz = trans[i*3+2]*0.1f;
  const float* o = Onv + (long)i * (NG * 48) + g * 48;
  _Float16* f = feat + (long)i * FEATW;
  for (int c = 0; c < PPG; ++c) f[1536 + g * PPG + c] = (_Float16)o[c];
  for (int p = 0; p < NPV; ++p) {
    const float vx = o[16 + p*3 + 0] - tx;
    const float vy = o[16 + p*3 + 1] - ty;
    const float vz = o[16 + p*3 + 2] - tz;
    const float ox = R[0]*vx + R[3]*vy + R[6]*vz;  // R^T
    const float oy = R[1]*vx + R[4]*vy + R[7]*vz;
    const float oz = R[2]*vx + R[5]*vy + R[8]*vz;
    f[1728 + g*24 + p*3 + 0] = (_Float16)ox;
    f[1728 + g*24 + p*3 + 1] = (_Float16)oy;
    f[1728 + g*24 + p*3 + 2] = (_Float16)oz;
    f[2016 + g*8 + p] = (_Float16)sqrtf(ox*ox + oy*oy + oz*oz + 1e-8f);
  }
}

// ---------------------------------------------------------------------------
extern "C" void kernel_launch(void* const* d_in, const int* in_sizes, int n_in,
                              void* d_out, int out_size, void* d_ws, size_t ws_size,
                              hipStream_t stream)
{
  const float* Node  = (const float*)d_in[0];
  const float* Edge  = (const float*)d_in[1];
  const float* rots  = (const float*)d_in[2];
  const float* trans = (const float*)d_in[3];
  const float* Wq    = (const float*)d_in[4];
  const float* Wk    = (const float*)d_in[5];
  const float* Wv    = (const float*)d_in[6];
  const float* Wqv   = (const float*)d_in[7];
  const float* Wkv   = (const float*)d_in[8];
  const float* Wvv   = (const float*)d_in[9];
  const float* Wbias = (const float*)d_in[10];
  const float* gamma = (const float*)d_in[11];
  const float* Wout  = (const float*)d_in[12];
  const float* bout  = (const float*)d_in[13];
  float* out = (float*)d_out;

  size_t off = 0;
  auto alloc = [&](size_t bytes) -> void* {
    void* p = (char*)d_ws + off;
    off += (bytes + 255) & ~(size_t)255;
    return p;
  };
  float*    cg     = (float*)alloc(64);
  float*    s2cg   = (float*)alloc(64);
  _Float16* Node_h = (_Float16*)alloc((size_t)LSEQ * CN * 2);
  _Float16* Wall_h = (_Float16*)alloc((size_t)CN * PW * 2);
  _Float16* Wbias_h= (_Float16*)alloc((size_t)CE * 16 * 2);
  _Float16* Wout_h = (_Float16*)alloc((size_t)FEATW * CN * 2);
  float*    P      = (float*)alloc((size_t)LSEQ * PW * 4);
  _Float16* QH     = (_Float16*)alloc((size_t)LSEQ * NG * 32 * 2);
  _Float16* KH     = (_Float16*)alloc((size_t)LSEQ * NG * 32 * 2);
  _Float16* VA     = (_Float16*)alloc((size_t)LSEQ * NG * 48 * 2);
  float*    sqnq   = (float*)alloc((size_t)LSEQ * NG * 4);
  float*    sqnk   = (float*)alloc((size_t)LSEQ * NG * 4);
  float*    S      = (float*)alloc((size_t)NG * PLANE * 4);
  _Float16* attn   = (_Float16*)alloc((size_t)16 * PLANE * 2); // 16 planes: padded M/N reads
  float*    Onv    = (float*)alloc((size_t)LSEQ * NG * 48 * 4);
  _Float16* feat   = (_Float16*)alloc((size_t)LSEQ * FEATW * 2);
  _Float16* Edge_t = (_Float16*)alloc((size_t)LSEQ * CE * LSEQ * 2); // [i][c][j] f16

  // --- prep ---
  calc_cg_k<<<1, 32, 0, stream>>>(gamma, cg, s2cg);
  cvt_k<<<768, 256, 0, stream>>>(Node, Node_h, LSEQ * CN);
  pack_wall_k<<<1024, 256, 0, stream>>>(Wq, Wk, Wv, Wqv, Wkv, Wvv, Wall_h);
  pack_wbias_k<<<8, 256, 0, stream>>>(Wbias, Wbias_h);
  cvt_k<<<1024, 256, 0, stream>>>(Wout, Wout_h, FEATW * CN);

  // --- projections: P[512,1152] = Node @ [Wq|Wk|Wv|Wqv|Wkv|Wvv] ---
  gemm_wmma_k<4, 2, _Float16, _Float16, float, false, false>
      <<<dim3(PW / 32, LSEQ / 64, 1), 256, 0, stream>>>(
      Node_h, CN, 0,  Wall_h, PW, 0,  P, PW, 1, 0,
      nullptr, LSEQ, PW, CN);

  frame_pack_k<<<(LSEQ * NG) / 256, 256, 0, stream>>>(
      P, rots, trans, cg, s2cg, QH, KH, VA, sqnq, sqnk);

  // --- Edge pass 1 (TDM): S[g,i,j] = Edge @ Wbias, plus Edge_t writeout ---
  bias_edge_tdm_k<<<dim3(LSEQ / 128, LSEQ, 1), 256, 0, stream>>>(
      Edge, Wbias_h, S, Edge_t);

  // --- qk + point cross-term: S[g,i,j] += qhat[i,g,:] . khat[j,g,:] (K=32) ---
  gemm_wmma_k<2, 4, _Float16, _Float16, float, true, true>
      <<<dim3(LSEQ / 64, LSEQ / 32, NG), 256, 0, stream>>>(
      QH, NG * 32, 32,  KH, NG * 32, 32,
      S, LSEQ, 1, PLANE,
      nullptr, LSEQ, LSEQ, 32);

  softmax_k<<<NG * LSEQ, 256, 0, stream>>>(S, sqnq, sqnk, attn);

  // --- Edge pass 2: feat[i, g*128+c] = sum_j Edge_t[i,c,j] * attn[g,i,j]
  //     (M=c, N=g, K=j; both operands K-contiguous f16) ---
  gemm_wmma_k<8, 1, _Float16, _Float16, _Float16, false, true>
      <<<dim3(1, 1, LSEQ), 256, 0, stream>>>(
      Edge_t, LSEQ, (long)CE * LSEQ,  attn, PLANE, LSEQ,
      feat, 1, CE, FEATW,
      nullptr, CE, NG, LSEQ);

  // --- O_Node + O_vec: Onv[i, g, 0:48] = sum_j attn[g,i,j] * VA[j,g,:] ---
  gemm_wmma_k<8, 1, _Float16, _Float16, float, false, false>
      <<<dim3(3, 4, NG), 256, 0, stream>>>(
      attn, LSEQ, PLANE,  VA, NG * 48, 48,
      Onv, NG * 48, 1, 48,
      nullptr, LSEQ, 48, LSEQ);

  finalize_k<<<(LSEQ * NG) / 256, 256, 0, stream>>>(Onv, rots, trans, feat);

  // --- output: out = feat @ Wout + bout ---
  gemm_wmma_k<4, 2, _Float16, _Float16, float, false, false>
      <<<dim3(CN / 32, LSEQ / 64, 1), 256, 0, stream>>>(
      feat, FEATW, 0,  Wout_h, CN, 0,  out, CN, 1, 0,
      bout, LSEQ, CN, FEATW);
}
